// TemporalAttention_21071109554596
// MI455X (gfx1250) — matrix operand order repaired and checked by reading, not compiled, not run on hardware
//
#include <hip/hip_runtime.h>
#include <hip/hip_bf16.h>

namespace {

constexpr int T_DIM = 64;   // sequence length
constexpr int H_DIM = 64;   // hidden / head dim
constexpr int N_HEADS = 8;
constexpr int STR  = 68;    // bf16 LDS row stride (elements); 136B row = 8B aligned
constexpr int SSTR = 68;    // f32 score row stride (elements); 272B row = 16B aligned
constexpr float INV_SQRT_H = 0.125f;                 // 1/sqrt(64)
constexpr float PE_LOG = 0.14391156831212787f;       // ln(10000)/64

typedef __attribute__((ext_vector_type(16))) __bf16 v16bf;
typedef __attribute__((ext_vector_type(8)))  float  v8f;

struct Lds {
  __bf16 h [T_DIM * STR];   // encoder output (row-major t,h)
  __bf16 w [H_DIM * STR];   // staged weight slice, N-major (n,k)
  __bf16 q [T_DIM * STR];   // q, later attn probs (row-major)
  __bf16 k [T_DIM * STR];   // k (row-major s,h), later o (row-major t,d)
  __bf16 vT[H_DIM * STR];   // v transposed (d,s) = B operand for attn@v
  float  sc[T_DIM * SSTR];  // x staging overlay (bf16), later f32 scores
};

__device__ __forceinline__ __bf16 bf(float f) { return (__bf16)f; }

// pack two floats into a dword of two bf16
__device__ __forceinline__ unsigned pk2(float a, float b) {
  union { unsigned u; __bf16 e[2]; } p;
  p.e[0] = bf(a); p.e[1] = bf(b);
  return p.u;
}

// A operand: 16x32 bf16 tile from LDS row-major (m,k).  ISA 7.12.2 layout.
__device__ __forceinline__ v16bf load_A(const __bf16* A, int mi, int kBase) {
  const int lane = threadIdx.x & 31;
  const int half = lane >> 4;
  const __bf16* rp = A + (mi * 16 + (lane & 15)) * STR + kBase;
  v16bf a;
#pragma unroll
  for (int j = 0; j < 8; ++j) {
    const int k = ((j & 4) << 2) + half * 8 + (j & 3) * 2;  // pair base K
    union { unsigned u; __bf16 e[2]; } p;
    p.u = *reinterpret_cast<const unsigned*>(rp + k);       // ds_load_b32
    a[2 * j]     = p.e[0];
    a[2 * j + 1] = p.e[1];
  }
  return a;
}

// B operand: 32x16 bf16 tile, LDS stored N-major (n,k) so K pairs contiguous.
__device__ __forceinline__ v16bf load_B(const __bf16* Bm, int ni, int kBase) {
  const int lane = threadIdx.x & 31;
  const int half = lane >> 4;
  const __bf16* rp = Bm + (ni * 16 + (lane & 15)) * STR + kBase;
  v16bf b;
#pragma unroll
  for (int j = 0; j < 8; ++j) {
    union { unsigned u; __bf16 e[2]; } p;
    p.u = *reinterpret_cast<const unsigned*>(rp + ((j & 4) << 2) + half * 8 + (j & 3) * 2);
    b[2 * j]     = p.e[0];
    b[2 * j + 1] = p.e[1];
  }
  return b;
}

// 64x64x64 GEMM: 8 waves, each wave owns two 16x16 f32 tiles.
// A fragments hoisted: reused across both N-tiles of the wave.
__device__ __forceinline__ void gemm64(const __bf16* A, const __bf16* Bm,
                                       v8f acc[2], bool accumulate) {
  const int wave = threadIdx.x >> 5;
  const int mi = wave >> 1;
  const v16bf a0 = load_A(A, mi, 0);
  const v16bf a1 = load_A(A, mi, 32);
#pragma unroll
  for (int t2 = 0; t2 < 2; ++t2) {
    const int ni = (wave & 1) * 2 + t2;
    v8f c = {0.f, 0.f, 0.f, 0.f, 0.f, 0.f, 0.f, 0.f};
    if (accumulate) c = acc[t2];
    const v16bf b0 = load_B(Bm, ni, 0);
    c = __builtin_amdgcn_wmma_f32_16x16x32_bf16(false, a0, false, b0,
                                                (short)0, c, false, false);
    const v16bf b1 = load_B(Bm, ni, 32);
    c = __builtin_amdgcn_wmma_f32_16x16x32_bf16(false, a1, false, b1,
                                                (short)0, c, false, false);
    acc[t2] = c;
  }
}

// C/D layout (ISA): VGPR r, lane l -> M = r + (l>=16 ? 8 : 0), N = l&15.
__device__ __forceinline__ void tile_coords(int t2, int& mb, int& n) {
  const int wave = threadIdx.x >> 5;
  const int lane = threadIdx.x & 31;
  n  = ((wave & 1) * 2 + t2) * 16 + (lane & 15);
  mb = (wave >> 1) * 16 + ((lane >> 4) << 3);
}

__device__ __forceinline__ void store_bf16(__bf16* dst, const v8f& c, int t2,
                                           const float* bias, float scale,
                                           bool transpose) {
  int mb, n; tile_coords(t2, mb, n);
  const float bv = bias ? bias[n] : 0.0f;
#pragma unroll
  for (int r = 0; r < 8; ++r) {
    const float v = (c[r] + bv) * scale;
    if (transpose) dst[n * STR + (mb + r)] = bf(v);
    else           dst[(mb + r) * STR + n] = bf(v);
  }
}

// h = gemm + b_enc + sinusoidal PE, computed on the fly.
__device__ __forceinline__ void store_h(__bf16* dst, const v8f& c, int t2,
                                        const float* b_enc) {
  int mb, n; tile_coords(t2, mb, n);
  const float bv = b_enc[n];
  const float freq = __expf(-(float)(n & 62) * PE_LOG);
#pragma unroll
  for (int r = 0; r < 8; ++r) {
    const int m = mb + r;
    const float ang = (float)m * freq;
    const float pe = (n & 1) ? __cosf(ang) : __sinf(ang);
    dst[m * STR + n] = bf(c[r] + bv + pe);
  }
}

__device__ __forceinline__ void store_scores(float* sc, const v8f& c, int t2) {
  int mb, n; tile_coords(t2, mb, n);
#pragma unroll
  for (int r = 0; r < 8; ++r) sc[(mb + r) * SSTR + n] = c[r];
}

// ---- software-pipelined weight staging (vectorized b128 loads) ---------
// wload4: 4 coalesced float4 global loads of a 64x64 f32 slice into regs.
// wcommit4: convert to bf16 and scatter into LDS N-major (n,k) layout.
__device__ __forceinline__ void wload4(float4* r, const float* __restrict__ W,
                                       int g) {
  const float4* __restrict__ W4 = reinterpret_cast<const float4*>(W);
  const int tid = threadIdx.x;
  const int g4 = g >> 2;
#pragma unroll
  for (int it = 0; it < 4; ++it) {
    const int idx = it * 256 + tid;          // float4 index in 64x16 grid
    r[it] = W4[(idx >> 4) * g4 + (idx & 15)];
  }
}
__device__ __forceinline__ void wcommit4(__bf16* dst, const float4* r) {
  const int tid = threadIdx.x;
#pragma unroll
  for (int it = 0; it < 4; ++it) {
    const int idx = it * 256 + tid;
    const int k = idx >> 4, n = (idx & 15) * 4;
    dst[(n + 0) * STR + k] = bf(r[it].x);
    dst[(n + 1) * STR + k] = bf(r[it].y);
    dst[(n + 2) * STR + k] = bf(r[it].z);
    dst[(n + 3) * STR + k] = bf(r[it].w);
  }
}

__global__ __launch_bounds__(256)
void ta_fused_kernel(const float* __restrict__ x,
                     const float* __restrict__ W_enc, const float* __restrict__ b_enc,
                     const float* __restrict__ Wq, const float* __restrict__ bq,
                     const float* __restrict__ Wk, const float* __restrict__ bk,
                     const float* __restrict__ Wv, const float* __restrict__ bv,
                     const float* __restrict__ Wo, const float* __restrict__ bo,
                     float* __restrict__ out, int N) {
  __shared__ Lds lds;
  const int tid = threadIdx.x;
  const int b  = blockIdx.x / N;
  const int st = blockIdx.x % N;

  float4 wreg[4];
  wload4(wreg, W_enc, H_DIM);             // W_enc loads in flight...

  // ---- stage x[b,:,st,:] (T x D) as bf16 (b128 loads, b64 LDS stores) ----
  {
    __bf16* xs = reinterpret_cast<__bf16*>(lds.sc);
    const float4* xb4 = reinterpret_cast<const float4*>(
        x + ((size_t)b * T_DIM * N + st) * H_DIM);
    const int rs4 = (N * H_DIM) >> 2;       // float4 row stride
#pragma unroll
    for (int it = 0; it < 4; ++it) {
      const int idx = it * 256 + tid;       // float4 index in 64x16 grid
      const int t = idx >> 4, c4 = idx & 15;
      const float4 v = xb4[(size_t)t * rs4 + c4];
      uint2 w2;
      w2.x = pk2(v.x, v.y);
      w2.y = pk2(v.z, v.w);
      *reinterpret_cast<uint2*>(xs + t * STR + c4 * 4) = w2;  // ds_store_b64
    }
  }
  wcommit4(lds.w, wreg);                  // ...landed; convert into LDS
  wload4(wreg, Wq, N_HEADS * H_DIM);      // prefetch Wq[0] under encoder GEMM
  __syncthreads();

  // ---- encoder: h = x @ W_enc + b_enc + PE ----
  {
    v8f d2[2];
    gemm64(reinterpret_cast<const __bf16*>(lds.sc), lds.w, d2, false);
    store_h(lds.h, d2[0], 0, b_enc);
    store_h(lds.h, d2[1], 1, b_enc);
  }
  __syncthreads();

  v8f out_acc[2] = {{0.f,0.f,0.f,0.f,0.f,0.f,0.f,0.f},
                    {0.f,0.f,0.f,0.f,0.f,0.f,0.f,0.f}};

  for (int hh = 0; hh < N_HEADS; ++hh) {
    v8f d2[2];
    // q = h @ Wq[:,hh] + bq   (pre-scaled by 1/sqrt(H))
    wcommit4(lds.w, wreg);                           // Wq[hh]
    wload4(wreg, Wk + hh * H_DIM, N_HEADS * H_DIM);  // prefetch Wk[hh]
    __syncthreads();
    gemm64(lds.h, lds.w, d2, false);
    store_bf16(lds.q, d2[0], 0, bq + hh * H_DIM, INV_SQRT_H, false);
    store_bf16(lds.q, d2[1], 1, bq + hh * H_DIM, INV_SQRT_H, false);
    __syncthreads();
    // k = h @ Wk[:,hh] + bk   (row-major (s,h) == N-major B for q·kT)
    wcommit4(lds.w, wreg);                           // Wk[hh]
    wload4(wreg, Wv + hh * H_DIM, N_HEADS * H_DIM);  // prefetch Wv[hh]
    __syncthreads();
    gemm64(lds.h, lds.w, d2, false);
    store_bf16(lds.k, d2[0], 0, bk + hh * H_DIM, 1.0f, false);
    store_bf16(lds.k, d2[1], 1, bk + hh * H_DIM, 1.0f, false);
    __syncthreads();
    // v = h @ Wv[:,hh] + bv   stored transposed (d,s) for attn·v
    wcommit4(lds.w, wreg);                           // Wv[hh]
    wload4(wreg, Wo + (size_t)hh * H_DIM * H_DIM, H_DIM); // prefetch Wo[hh]
    __syncthreads();
    gemm64(lds.h, lds.w, d2, false);
    store_bf16(lds.vT, d2[0], 0, bv + hh * H_DIM, 1.0f, true);
    store_bf16(lds.vT, d2[1], 1, bv + hh * H_DIM, 1.0f, true);
    __syncthreads();
    // scores = q @ kT (scale folded into q), f32 into LDS
    gemm64(lds.q, lds.k, d2, false);
    store_scores(lds.sc, d2[0], 0);
    store_scores(lds.sc, d2[1], 1);
    __syncthreads();
    // causal softmax, fully parallel: 4 threads/row, shuffle reduction,
    // b128 score reads + packed b64 attn writes.
    {
      const int row  = tid >> 2;          // 0..63
      const int part = tid & 3;           // 16-element chunk
      const float4* srow4 =
          reinterpret_cast<const float4*>(lds.sc + row * SSTR + part * 16);
      float vbuf[16];
#pragma unroll
      for (int i4 = 0; i4 < 4; ++i4) {
        const float4 v4 = srow4[i4];      // ds_load_b128
        vbuf[4 * i4 + 0] = v4.x; vbuf[4 * i4 + 1] = v4.y;
        vbuf[4 * i4 + 2] = v4.z; vbuf[4 * i4 + 3] = v4.w;
      }
      float mx = -3.0e38f;
#pragma unroll
      for (int i = 0; i < 16; ++i) {
        const int s = part * 16 + i;
        vbuf[i] = (s <= row) ? vbuf[i] : -3.0e38f;
        mx = fmaxf(mx, vbuf[i]);
      }
      mx = fmaxf(mx, __shfl_xor(mx, 1, 4));
      mx = fmaxf(mx, __shfl_xor(mx, 2, 4));
      float sum = 0.0f;
#pragma unroll
      for (int i = 0; i < 16; ++i) {
        const int s = part * 16 + i;
        const float e = (s <= row) ? __expf(vbuf[i] - mx) : 0.0f;
        vbuf[i] = e;
        sum += e;
      }
      sum += __shfl_xor(sum, 1, 4);
      sum += __shfl_xor(sum, 2, 4);
      const float inv = 1.0f / sum;
      uint2* arow2 =
          reinterpret_cast<uint2*>(lds.q + row * STR + part * 16);
#pragma unroll
      for (int i4 = 0; i4 < 4; ++i4) {
        uint2 w2;
        w2.x = pk2(vbuf[4 * i4 + 0] * inv, vbuf[4 * i4 + 1] * inv);
        w2.y = pk2(vbuf[4 * i4 + 2] * inv, vbuf[4 * i4 + 3] * inv);
        arow2[i4] = w2;                   // ds_store_b64
      }
    }
    __syncthreads();
    // o = attn @ v  -> reuse the dead k buffer (row-major (t,d))
    gemm64(lds.q, lds.vT, d2, false);
    store_bf16(lds.k, d2[0], 0, nullptr, 1.0f, false);
    store_bf16(lds.k, d2[1], 1, nullptr, 1.0f, false);
    __syncthreads();
    // out += o @ Wo[hh]   — accumulated in WMMA C registers
    wcommit4(lds.w, wreg);                           // Wo[hh]
    // prefetch next head's Wq (harmless dup for the last head)
    wload4(wreg, Wq + ((hh + 1) & (N_HEADS - 1)) * H_DIM, N_HEADS * H_DIM);
    __syncthreads();
    gemm64(lds.k, lds.w, out_acc, hh > 0);
    __syncthreads();
  }

  // ---- epilogue: out = acc + bo ----
  {
    float* ob = out + ((size_t)b * N + st) * T_DIM * H_DIM;
#pragma unroll
    for (int t2 = 0; t2 < 2; ++t2) {
      int mb, n; tile_coords(t2, mb, n);
      const float bv = bo[n];
#pragma unroll
      for (int r = 0; r < 8; ++r)
        ob[(size_t)(mb + r) * H_DIM + n] = out_acc[t2][r] + bv;
    }
  }
}

} // namespace

extern "C" void kernel_launch(void* const* d_in, const int* in_sizes, int n_in,
                              void* d_out, int out_size, void* d_ws, size_t ws_size,
                              hipStream_t stream) {
  (void)n_in; (void)out_size; (void)d_ws; (void)ws_size;
  const float* x     = (const float*)d_in[0];
  const float* W_enc = (const float*)d_in[1];
  const float* b_enc = (const float*)d_in[2];
  const float* Wq    = (const float*)d_in[3];
  const float* bq    = (const float*)d_in[4];
  const float* Wk    = (const float*)d_in[5];
  const float* bk    = (const float*)d_in[6];
  const float* Wv    = (const float*)d_in[7];
  const float* bv    = (const float*)d_in[8];
  const float* Wo    = (const float*)d_in[9];
  const float* bo    = (const float*)d_in[10];
  float* out = (float*)d_out;

  const int N  = 200;                       // stations (fixed by reference)
  const int BN = in_sizes[0] / (64 * 64);   // B*N = 800

  ta_fused_kernel<<<dim3(BN), dim3(256), 0, stream>>>(
      x, W_enc, b_enc, Wq, bq, Wk, bk, Wv, bv, Wo, bo, out, N);
}
